// NLGRNN_6700148982017
// MI455X (gfx1250) — matrix operand order repaired and checked by reading, not compiled, not run on hardware
//
#include <hip/hip_runtime.h>
#include <hip/hip_bf16.h>
#include <math.h>

#define DEV __device__ __forceinline__

typedef __attribute__((ext_vector_type(16))) __bf16 v16bf;
typedef __attribute__((ext_vector_type(8)))  __bf16 bf8v;
typedef __attribute__((ext_vector_type(8)))  float  v8f;
typedef __attribute__((ext_vector_type(4)))  float  f4v;   // native vec for NT stores

constexpr int B_ = 8, K_ = 5, T_ = 40, V_ = 32000, E_ = 256, H_ = 512, A_ = 512;
constexpr int KB  = K_ * B_;      // 40 rows (beam*batch)
constexpr int G3  = 3 * H_;       // 1536 gate width
constexpr float NEGINF = -1e9f;

// ---------------- WMMA helpers (CDNA5 bf16 16x16x32) ----------------

DEV v8f zero8() {
  v8f c;
  #pragma unroll
  for (int i = 0; i < 8; ++i) c[i] = 0.f;
  return c;
}

DEV v8f wmma_bf16(v16bf a, v16bf b, v8f c) {
  // D = A*B + C ; emits v_wmma_f32_16x16x32_bf16
  return __builtin_amdgcn_wmma_f32_16x16x32_bf16(false, a, false, b,
                                                 (short)0, c, false, false);
}

// A-matrix 16x32 bf16: lane = row (mod 16); lane<16 holds K 0..7,16..23 ;
// lane>=16 holds K 8..15,24..31 (ISA 7.12.2). Rows >= RMAX are zero-padded.
DEV v16bf load_a_bf16(const __bf16* P, int r0, int RMAX, int ld, int k0, int lane) {
  int r  = r0 + (lane & 15);
  int hi = lane >> 4;
  v16bf a;
  if (r < RMAX) {
    const __bf16* p = P + (size_t)r * ld + k0 + hi * 8;
    bf8v lo  = *(const bf8v*)(p);
    bf8v hi8 = *(const bf8v*)(p + 16);
    #pragma unroll
    for (int i = 0; i < 8; ++i) { a[i] = lo[i]; a[i + 8] = hi8[i]; }
  } else {
    #pragma unroll
    for (int i = 0; i < 16; ++i) a[i] = (__bf16)0.f;
  }
  return a;
}

// Same layout but converting from an fp32 source (hidden state kept fp32).
DEV v16bf load_a_f32(const float* P, int r0, int RMAX, int ld, int k0, int lane) {
  int r  = r0 + (lane & 15);
  int hi = lane >> 4;
  v16bf a;
  if (r < RMAX) {
    const float* p = P + (size_t)r * ld + k0 + hi * 8;
    #pragma unroll
    for (int i = 0; i < 8; ++i) { a[i] = (__bf16)p[i]; a[i + 8] = (__bf16)p[16 + i]; }
  } else {
    #pragma unroll
    for (int i = 0; i < 16; ++i) a[i] = (__bf16)0.f;
  }
  return a;
}

// B-matrix 32x16 bf16: lane = column; lane<16 holds K 0..7,16..23, lane>=16 K 8..15,24..31.
// W is row-major [N][Kd] (the transposed weight as stored), so B[k][n]=W[n][k].
DEV v16bf load_b_bf16(const __bf16* W, int n0, int ld, int k0, int lane) {
  int n  = n0 + (lane & 15);
  int kk = k0 + (lane >> 4) * 8;
  const __bf16* p = W + (size_t)n * ld + kk;
  bf8v lo  = *(const bf8v*)(p);
  bf8v hi8 = *(const bf8v*)(p + 16);
  v16bf b;
  #pragma unroll
  for (int i = 0; i < 8; ++i) { b[i] = lo[i]; b[i + 8] = hi8[i]; }
  return b;
}

// ---------------- small prep kernels ----------------

__global__ void k_conv_bf16(const float* __restrict__ s, __bf16* __restrict__ d, int n) {
  int i = blockIdx.x * blockDim.x + threadIdx.x;
  if (i < n) d[i] = (__bf16)s[i];
}

// h0 = attrs @ transform_w.T + transform_b, broadcast over beams; init bp/last_ids
__global__ void k_init(const float* __restrict__ attrs, const float* __restrict__ tw,
                       const float* __restrict__ tb, const int* __restrict__ bos,
                       float* __restrict__ h_cur, float* __restrict__ bp,
                       int* __restrict__ last_ids) {
  const int b = blockIdx.x;
  for (int j = threadIdx.x; j < H_; j += blockDim.x) {
    float acc = tb[j];
    for (int a = 0; a < A_; ++a) acc += attrs[b * A_ + a] * tw[j * A_ + a];
    for (int k = 0; k < K_; ++k) h_cur[(k * B_ + b) * H_ + j] = acc;
  }
  if (threadIdx.x < K_) {
    bp[threadIdx.x * B_ + b] = (threadIdx.x == 0) ? 0.f : NEGINF;
    last_ids[threadIdx.x * B_ + b] = bos[0];
  }
}

__global__ void k_embed(const float* __restrict__ emb, const int* __restrict__ last_ids,
                        __bf16* __restrict__ x) {
  int kb = blockIdx.x;
  int id = last_ids[kb];
  for (int e = threadIdx.x; e < E_; e += blockDim.x)
    x[kb * E_ + e] = (__bf16)emb[(size_t)id * E_ + e];
}

// ---------------- fused GRU cell: 6 WMMA tile-GEMMs + gates ----------------
// grid (H/16, ceil(KB/16)), block 32 (one wave)
__global__ void k_gru(const __bf16* __restrict__ x,   const __bf16* __restrict__ wih,
                      const __bf16* __restrict__ whh, const float* __restrict__ bih,
                      const float* __restrict__ bhh,  const float* __restrict__ h_cur,
                      float* __restrict__ h_next,     __bf16* __restrict__ hn_bf) {
  const int lane = threadIdx.x & 31;
  const int jt   = blockIdx.x;        // 16-wide slice of H
  const int row0 = blockIdx.y * 16;   // row tile over KB
  v8f gir = zero8(), giz = zero8(), gin = zero8();
  v8f ghr = zero8(), ghz = zero8(), ghn = zero8();
  // gi = x @ w_ih^T  (A=[KB,E], W=[3H,E])
  for (int k0 = 0; k0 < E_; k0 += 32) {
    v16bf a  = load_a_bf16(x, row0, KB, E_, k0, lane);
    v16bf b0 = load_b_bf16(wih, jt * 16,          E_, k0, lane);
    v16bf b1 = load_b_bf16(wih, H_ + jt * 16,     E_, k0, lane);
    v16bf b2 = load_b_bf16(wih, 2 * H_ + jt * 16, E_, k0, lane);
    gir = wmma_bf16(a, b0, gir); giz = wmma_bf16(a, b1, giz); gin = wmma_bf16(a, b2, gin);
  }
  // gh = h @ w_hh^T  (A=[KB,H] fp32 src, W=[3H,H])
  for (int k0 = 0; k0 < H_; k0 += 32) {
    v16bf a  = load_a_f32(h_cur, row0, KB, H_, k0, lane);
    v16bf b0 = load_b_bf16(whh, jt * 16,          H_, k0, lane);
    v16bf b1 = load_b_bf16(whh, H_ + jt * 16,     H_, k0, lane);
    v16bf b2 = load_b_bf16(whh, 2 * H_ + jt * 16, H_, k0, lane);
    ghr = wmma_bf16(a, b0, ghr); ghz = wmma_bf16(a, b1, ghz); ghn = wmma_bf16(a, b2, ghn);
  }
  // gates: D layout -> lane holds col n=lane&15, rows m=(lane>>4)*8+r
  const int n  = lane & 15;
  const int j  = jt * 16 + n;
  const int mb = row0 + (lane >> 4) * 8;
  const float bir = bih[j], biz = bih[H_ + j], bin = bih[2 * H_ + j];
  const float bhr = bhh[j], bhz = bhh[H_ + j], bhn = bhh[2 * H_ + j];
  #pragma unroll
  for (int r = 0; r < 8; ++r) {
    int m = mb + r;
    if (m < KB) {
      float hprev = h_cur[m * H_ + j];
      float rg = 1.f / (1.f + __expf(-(gir[r] + bir + ghr[r] + bhr)));
      float zg = 1.f / (1.f + __expf(-(giz[r] + biz + ghz[r] + bhz)));
      float ng = tanhf(gin[r] + bin + rg * (ghn[r] + bhn));
      float hnew = (1.f - zg) * ng + zg * hprev;
      h_next[m * H_ + j] = hnew;
      hn_bf[m * H_ + j]  = (__bf16)hnew;
    }
  }
}

// ---- bf16 WMMA GEMM, RT row-tiles per wave (B-tile reuse + acc interleave) ----
// out[M,N] = A[M,Kd] @ W[N,Kd]^T + bias
// grid (N/16, ceil(M/(16*RT))), block 32 (one wave). Kd%32==0, N%16==0.
template <int RT, bool NTSTORE>
__global__ void k_gemm(const __bf16* __restrict__ A, int M,
                       const __bf16* __restrict__ W, const float* __restrict__ bias,
                       float* __restrict__ out, int N, int Kd) {
  const int lane = threadIdx.x & 31;
  const int n0   = blockIdx.x * 16;
  const int row0 = blockIdx.y * 16 * RT;
  v8f acc[RT];
  #pragma unroll
  for (int r = 0; r < RT; ++r) acc[r] = zero8();
  for (int k0 = 0; k0 < Kd; k0 += 32) {
    v16bf b = load_b_bf16(W, n0, Kd, k0, lane);
    // stream-ahead hint for the B matrix (emits global_prefetch_b8)
    __builtin_prefetch(W + (size_t)(n0 + (lane & 15)) * Kd + k0 + 64, 0, 1);
    #pragma unroll
    for (int r = 0; r < RT; ++r) {
      v16bf a = load_a_bf16(A, row0 + 16 * r, M, Kd, k0, lane);
      acc[r] = wmma_bf16(a, b, acc[r]);   // independent accumulators interleave
    }
  }
  const int n  = n0 + (lane & 15);
  const float bs = bias[n];
  #pragma unroll
  for (int r = 0; r < RT; ++r) {
    const int mb = row0 + 16 * r + (lane >> 4) * 8;
    #pragma unroll
    for (int q = 0; q < 8; ++q) {
      int m = mb + q;
      if (m < M) {
        float val = acc[r][q] + bs;
        if (NTSTORE) __builtin_nontemporal_store(val, &out[(size_t)m * N + n]);
        else         out[(size_t)m * N + n] = val;
      }
    }
  }
}

// ---------------- fused logsumexp + top-5 per row (40 rows x 32000) ----------------
__global__ void k_topk(const float* __restrict__ out_buf,
                       float* __restrict__ tp, int* __restrict__ ti) {
  const int row = blockIdx.x, tid = threadIdx.x;
  const float* src = out_buf + (size_t)row * V_;
  float lv[5]; int li[5];
  #pragma unroll
  for (int i = 0; i < 5; ++i) { lv[i] = -3e38f; li[i] = 0; }
  float m = -3e38f, s = 0.f;
  for (int v = tid; v < V_; v += 256) {
    float x = src[v];
    if (x > m) { s = s * __expf(m - x) + 1.f; m = x; } else s += __expf(x - m);
    if (x > lv[4]) {
      int p = 4;
      while (p > 0 && lv[p - 1] < x) { lv[p] = lv[p - 1]; li[p] = li[p - 1]; --p; }
      lv[p] = x; li[p] = v;
    }
  }
  __shared__ float sv[256 * 5]; __shared__ int si[256 * 5];
  __shared__ float sm[256], ssum[256];
  #pragma unroll
  for (int i = 0; i < 5; ++i) { sv[tid * 5 + i] = lv[i]; si[tid * 5 + i] = li[i]; }
  sm[tid] = m; ssum[tid] = s;
  __syncthreads();
  for (int str = 128; str > 0; str >>= 1) {
    if (tid < str) {
      float m1 = sm[tid], s1 = ssum[tid], m2 = sm[tid + str], s2 = ssum[tid + str];
      float mm = fmaxf(m1, m2);
      ssum[tid] = s1 * __expf(m1 - mm) + s2 * __expf(m2 - mm);
      sm[tid] = mm;
      float av[5], bv[5]; int ai[5], bi[5];
      #pragma unroll
      for (int i = 0; i < 5; ++i) {
        av[i] = sv[tid * 5 + i];        ai[i] = si[tid * 5 + i];
        bv[i] = sv[(tid + str) * 5 + i]; bi[i] = si[(tid + str) * 5 + i];
      }
      int ia = 0, ib = 0;
      #pragma unroll
      for (int o = 0; o < 5; ++o) {
        if (av[ia] >= bv[ib]) { sv[tid * 5 + o] = av[ia]; si[tid * 5 + o] = ai[ia]; ++ia; }
        else                  { sv[tid * 5 + o] = bv[ib]; si[tid * 5 + o] = bi[ib]; ++ib; }
      }
    }
    __syncthreads();
  }
  if (tid == 0) {
    float lse = sm[0] + __logf(ssum[0]);
    for (int i = 0; i < 5; ++i) { tp[row * 5 + i] = sv[i] - lse; ti[row * 5 + i] = si[i]; }
  }
}

// ---------------- beam selection + hidden reorder ----------------
__global__ void k_beam(const float* __restrict__ tp, const int* __restrict__ ti,
                       float* __restrict__ bp, int* __restrict__ last_ids,
                       const float* __restrict__ h_next, float* __restrict__ h_cur,
                       int* __restrict__ org_t, int* __restrict__ tok_t) {
  __shared__ int s_org[B_ * K_];
  const int tid = threadIdx.x;
  if (tid < B_) {
    const int b = tid;
    float cand[K_ * K_]; int ctok[K_ * K_];
    for (int k = 0; k < K_; ++k) {
      float bpk = bp[k * B_ + b];
      for (int r = 0; r < K_; ++r) {
        cand[k * K_ + r] = bpk + tp[(k * B_ + b) * K_ + r];
        ctok[k * K_ + r] = ti[(k * B_ + b) * K_ + r];
      }
    }
    bool used[K_ * K_];
    for (int i = 0; i < K_ * K_; ++i) used[i] = false;
    for (int s = 0; s < K_; ++s) {
      int best = 0; float bv = -3.4e38f;
      for (int i = 0; i < K_ * K_; ++i)
        if (!used[i] && cand[i] > bv) { bv = cand[i]; best = i; }
      used[best] = true;
      int org = best / K_, tok = ctok[best];
      s_org[b * K_ + s] = org;
      org_t[b * K_ + s] = org;
      tok_t[b * K_ + s] = tok;
      bp[s * B_ + b] = bv;
      last_ids[s * B_ + b] = tok;
    }
  }
  __syncthreads();
  for (int e = tid; e < KB * H_; e += blockDim.x) {
    int srow = e / H_, j = e % H_;
    int s = srow / B_, b = srow % B_;
    h_cur[e] = h_next[(s_org[b * K_ + s] * B_ + b) * H_ + j];
  }
}

// ---------------- backtracking + output assembly ----------------
__global__ void k_backtrack(const int* __restrict__ org_all, const int* __restrict__ tok_all,
                            int* __restrict__ sel, int* __restrict__ tokc) {
  int tid = threadIdx.x;
  if (tid >= B_ * K_) return;
  int b = tid / K_, k = tid % K_;
  int idx = k;
  for (int t = T_ - 1; t >= 0; --t) {
    int o  = org_all[t * B_ * K_ + b * K_ + idx];
    int tk = tok_all[t * B_ * K_ + b * K_ + idx];
    sel [t * B_ * K_ + b * K_ + k] = o;   // beam into hn_store[t]
    tokc[t * B_ * K_ + b * K_ + k] = tk;
    idx = o;
  }
}

__global__ void k_gather(const __bf16* __restrict__ hn_store, const int* __restrict__ sel,
                         __bf16* __restrict__ Afin) {
  int R = blockIdx.x;                   // (b*K + k)*T + t
  int b = R / (K_ * T_), k = (R / T_) % K_, t = R % T_;
  int org = sel[t * B_ * K_ + b * K_ + k];
  const __bf16* src = hn_store + ((size_t)t * KB + org * B_ + b) * H_;
  __bf16* dst = Afin + (size_t)R * H_;
  for (int j = threadIdx.x; j < H_; j += blockDim.x) dst[j] = src[j];
}

__global__ void k_zero4(f4v* __restrict__ p, long n4) {
  long i = (long)blockIdx.x * blockDim.x + threadIdx.x;
  if (i < n4) {
    f4v z = {0.f, 0.f, 0.f, 0.f};
    __builtin_nontemporal_store(z, &p[i]);
  }
}

// outputs == y_hard numerically (straight-through): scatter one-hot ones
__global__ void k_ones(const int* __restrict__ tokc,
                       float* __restrict__ outp, float* __restrict__ yh) {
  int idx = blockIdx.x * blockDim.x + threadIdx.x;
  if (idx >= B_ * K_ * T_) return;
  int b = idx / (K_ * T_), k = (idx / T_) % K_, t = idx % T_;
  int tok = tokc[t * B_ * K_ + b * K_ + k];
  size_t off = (size_t)idx * V_ + tok;
  outp[off] = 1.f; yh[off] = 1.f;
}

// ---------------- host orchestration ----------------
extern "C" void kernel_launch(void* const* d_in, const int* in_sizes, int n_in,
                              void* d_out, int out_size, void* d_ws, size_t ws_size,
                              hipStream_t stream) {
  (void)in_sizes; (void)n_in; (void)out_size; (void)ws_size;
  const float* attrs = (const float*)d_in[0];
  const int*   bos   = (const int*)  d_in[1];
  const float* emb   = (const float*)d_in[4];
  const float* w_ih  = (const float*)d_in[5];
  const float* b_ih  = (const float*)d_in[6];
  const float* w_hh  = (const float*)d_in[7];
  const float* b_hh  = (const float*)d_in[8];
  const float* tw    = (const float*)d_in[9];
  const float* tb    = (const float*)d_in[10];
  const float* lw    = (const float*)d_in[11];
  const float* lb    = (const float*)d_in[12];
  float* out = (float*)d_out;

  // workspace carve-out (~44 MB total), 256B aligned
  char* w = (char*)d_ws;
  auto alloc = [&](size_t bytes) { char* p = w; w += (bytes + 255) & ~(size_t)255; return p; };
  __bf16* lin_bf   = (__bf16*)alloc((size_t)V_ * H_ * 2);
  __bf16* wih_bf   = (__bf16*)alloc((size_t)G3 * E_ * 2);
  __bf16* whh_bf   = (__bf16*)alloc((size_t)G3 * H_ * 2);
  __bf16* x_bf     = (__bf16*)alloc((size_t)KB * E_ * 2);
  __bf16* hn_store = (__bf16*)alloc((size_t)T_ * KB * H_ * 2);
  __bf16* Afin     = (__bf16*)alloc((size_t)B_ * K_ * T_ * H_ * 2);
  float* h_cur   = (float*)alloc((size_t)KB * H_ * 4);
  float* h_next  = (float*)alloc((size_t)KB * H_ * 4);
  float* out_buf = (float*)alloc((size_t)KB * V_ * 4);
  float* tp      = (float*)alloc(KB * K_ * 4);
  int*   ti      = (int*)  alloc(KB * K_ * 4);
  float* bp      = (float*)alloc(KB * 4);
  int*   last_ids= (int*)  alloc(KB * 4);
  int*   org_all = (int*)  alloc(T_ * B_ * K_ * 4);
  int*   tok_all = (int*)  alloc(T_ * B_ * K_ * 4);
  int*   sel     = (int*)  alloc(T_ * B_ * K_ * 4);
  int*   tokc    = (int*)  alloc(T_ * B_ * K_ * 4);

  // one-time bf16 weight conversion (linear_w lands in the 192MB L2 and stays)
  k_conv_bf16<<<(V_ * H_ + 255) / 256, 256, 0, stream>>>(lw, lin_bf, V_ * H_);
  k_conv_bf16<<<(G3 * E_ + 255) / 256, 256, 0, stream>>>(w_ih, wih_bf, G3 * E_);
  k_conv_bf16<<<(G3 * H_ + 255) / 256, 256, 0, stream>>>(w_hh, whh_bf, G3 * H_);
  k_init<<<B_, 256, 0, stream>>>(attrs, tw, tb, bos, h_cur, bp, last_ids);

  for (int t = 0; t < T_; ++t) {
    __bf16* hn_t = hn_store + (size_t)t * KB * H_;
    k_embed<<<KB, 64, 0, stream>>>(emb, last_ids, x_bf);
    k_gru<<<dim3(H_ / 16, (KB + 15) / 16), 32, 0, stream>>>(
        x_bf, wih_bf, whh_bf, b_ih, b_hh, h_cur, h_next, hn_t);
    // 3 row tiles (rows 0..47, M=40 guarded) share each streamed B tile
    k_gemm<3, false><<<dim3(V_ / 16, 1), 32, 0, stream>>>(
        hn_t, KB, lin_bf, lb, out_buf, V_, H_);
    k_topk<<<KB, 256, 0, stream>>>(out_buf, tp, ti);
    k_beam<<<1, 256, 0, stream>>>(tp, ti, bp, last_ids, h_next, h_cur,
                                  org_all + t * B_ * K_, tok_all + t * B_ * K_);
  }

  k_backtrack<<<1, 64, 0, stream>>>(org_all, tok_all, sel, tokc);
  k_gather<<<B_ * K_ * T_, 64, 0, stream>>>(hn_store, sel, Afin);
  // final logits [B,K,T,V]: 4 row tiles per wave, NT stores straight to HBM
  k_gemm<4, true><<<dim3(V_ / 16, (B_ * K_ * T_) / 64), 32, 0, stream>>>(
      Afin, B_ * K_ * T_, lin_bf, lb, out, V_, H_);
  // outputs & y_hard sections: zero (NT) then one-hot scatter
  size_t NBV = (size_t)B_ * K_ * T_ * V_;
  long n4 = (long)((2 * NBV) / 4);
  k_zero4<<<(unsigned)((n4 + 255) / 256), 256, 0, stream>>>((f4v*)(out + NBV), n4);
  k_ones<<<(B_ * K_ * T_ + 255) / 256, 256, 0, stream>>>(tokc, out + NBV, out + 2 * NBV);
}
